// KernelAttentionRegression_3289944949281
// MI455X (gfx1250) — matrix operand order repaired and checked by reading, not compiled, not run on hardware
//
#include <hip/hip_runtime.h>
#include <stdint.h>

// Problem constants (match reference)
#define S_    2048
#define D_    1024
#define H_    16
#define DH_   64
#define RK_   16
#define ITERS_ 5

// K-chunk of the B operand staged in LDS for the K*alpha GEMMs
#define KC_   512
#define KCP_  (KC_ + 8)     // +16B row pad: row stride 1040B -> 4-bank shift/row, conflict-free b128 reads

typedef __attribute__((ext_vector_type(16))) __bf16 v16bf;
typedef __attribute__((ext_vector_type(8)))  float  v8f;

// ---------- small helpers ----------
__device__ __forceinline__ uint16_t f2bf(float f) {
  uint32_t u = __builtin_bit_cast(uint32_t, f);
  u += 0x7FFFu + ((u >> 16) & 1u);           // round-to-nearest-even
  return (uint16_t)(u >> 16);
}
__device__ __forceinline__ float bf2f(uint16_t h) {
  uint32_t u = ((uint32_t)h) << 16;
  return __builtin_bit_cast(float, u);
}

struct __align__(16) U128 { uint32_t x[4]; };
union FragBF { U128 q[2]; v16bf v; };

// ---------------------------------------------------------------------------
// Core: one wave computes a 16x16 f32 tile of C = A (MxK, bf16 row-major,
// contiguous in K) times B^T (N x K, bf16 row-major, contiguous in K) with
// v_wmma_f32_16x16x32_bf16.
//
// Fragment layouts per CDNA5 ISA 7.12.2 (wave32):
//   A 16x32 : lanes 0-15 -> M=lane, K = {0..7, 16..23}; lanes 16-31 -> M=lane-16,
//             K = {8..15, 24..31}. Elements 0..7 then 8..15 of v16bf.
//   B 32x16 : lanes 0-15 -> N=lane, K = {0..7, 16..23}; lanes 16-31 -> N=lane-16, K = {8..15, 24..31}.
//   C/D     : VGPR r -> row r + 8*(lane>=16), col = lane&15.
// ---------------------------------------------------------------------------
__device__ __forceinline__ v8f wmma_acc_bt(const uint16_t* __restrict__ A, int lda,
                                           const uint16_t* __restrict__ BT, int ldb,
                                           int m0, int n0, int Kd)
{
  const int lane = threadIdx.x & 31;
  const int half = lane >> 4;
  const int l15  = lane & 15;
  const uint16_t* ap = A  + (size_t)(m0 + l15) * (size_t)lda + half * 8;
  const uint16_t* bp = BT + (size_t)(n0 + l15) * (size_t)ldb + half * 8;
  v8f c = {};
#pragma unroll 2
  for (int k0 = 0; k0 < Kd; k0 += 32) {
    FragBF a, b;
    a.q[0] = *(const U128*)(ap + k0);        // A[m][kb+0..7]
    a.q[1] = *(const U128*)(ap + k0 + 16);   // A[m][kb+16..23]
    b.q[0] = *(const U128*)(bp + k0);        // B^T[n][kb+0..7]
    b.q[1] = *(const U128*)(bp + k0 + 16);   // B^T[n][kb+16..23]
    c = __builtin_amdgcn_wmma_f32_16x16x32_bf16(false, a.v, false, b.v,
                                                (short)0, c, false, false);
  }
  return c;
}

// ---------------------------------------------------------------------------
// Wide K*alpha accumulator: 8-wave workgroup, each wave owns a 16-row strip
// and all 64 alpha columns (4 accumulators -> 4x A reuse, 4 WMMA per A frag).
// The shared B operand (alpha^T, 64 x 2048 bf16) is staged chunk-by-chunk into
// LDS with gfx1250 async global->LDS b128 copies (ASYNCcnt), then consumed via
// ds_load_b128 fragment reads while the A operand (K matrix) streams from L2.
// ---------------------------------------------------------------------------
__device__ __forceinline__ void kalpha_acc64(const uint16_t* __restrict__ Kh,
                                             const uint16_t* __restrict__ aTh,
                                             uint16_t* __restrict__ sB,
                                             v8f c[4])
{
  const int tid  = threadIdx.x;
  const int wv   = tid >> 5;
  const int lane = tid & 31;
  const int half = lane >> 4;
  const int l15  = lane & 15;
  const int m0   = blockIdx.y * 128 + wv * 16;
  const uint16_t* ap = Kh + (size_t)(m0 + l15) * S_ + half * 8;

  c[0] = (v8f){}; c[1] = (v8f){}; c[2] = (v8f){}; c[3] = (v8f){};

  for (int kb = 0; kb < S_; kb += KC_) {
    // --- async stage: aTh[row][kb..kb+KC) -> sB[row*KCP_ ...], 64 rows ---
    // 64*KC_/8 = 4096 b128 segments, 16 per thread, coalesced across lanes.
#pragma unroll
    for (int i = 0; i < (64 * KC_ / 8) / 256; ++i) {
      int g   = i * 256 + tid;
      int row = g >> 6;                 // KC_/8 = 64 segments per row
      int c8  = (g & 63) * 8;
      const uint16_t* src = aTh + (size_t)row * S_ + kb + c8;
      uint32_t dst = (uint32_t)(uintptr_t)(sB + row * KCP_ + c8);
      asm volatile("global_load_async_to_lds_b128 %0, %1, off"
                   :: "v"(dst), "v"((uint64_t)(uintptr_t)src) : "memory");
    }
    asm volatile("s_wait_asynccnt 0x0" ::: "memory");
    __syncthreads();

    // prefetch the next A chunk while consuming this one
    if (kb + KC_ < S_) __builtin_prefetch(ap + kb + KC_, 0, 1);

    // --- consume chunk: per k-step 1 A frag (global/L2) x 4 B frags (LDS) ---
#pragma unroll 4
    for (int k0 = 0; k0 < KC_; k0 += 32) {
      FragBF a;
      a.q[0] = *(const U128*)(ap + kb + k0);
      a.q[1] = *(const U128*)(ap + kb + k0 + 16);
#pragma unroll
      for (int nt = 0; nt < 4; ++nt) {
        FragBF b;
        const uint16_t* bp = sB + (nt * 16 + l15) * KCP_ + half * 8 + k0;
        b.q[0] = *(const U128*)(bp);
        b.q[1] = *(const U128*)(bp + 16);
        c[nt] = __builtin_amdgcn_wmma_f32_16x16x32_bf16(false, a.v, false, b.v,
                                                        (short)0, c[nt], false, false);
      }
    }
    __syncthreads();
  }
}

// ---------- scalar prep: softplus(bandwidth), softplus(lambda) ----------
__global__ void k_scalars(const float* __restrict__ bw_in,
                          const float* __restrict__ lam_in,
                          float* __restrict__ scal)
{
  if (threadIdx.x == 0 && blockIdx.x == 0) {
    float bw = log1pf(expf(bw_in[0])) + 1e-6f;
    scal[0] = 1.0f / (2.0f * bw * bw);       // multiplier inside exp
    scal[1] = log1pf(expf(lam_in[0]));       // lambda
  }
}

// ---------- f32 -> bf16 conversions ----------
__global__ void k_cvt(const float* __restrict__ in, uint16_t* __restrict__ out, int n)
{
  int t = blockIdx.x * 256 + threadIdx.x;
  if (t < n) out[t] = f2bf(in[t]);
}

// transpose-convert a DxD weight: out[n*D + k] = bf16(in[k*D + n])  (B^T GEMM path)
__global__ void k_cvtT(const float* __restrict__ in, uint16_t* __restrict__ out)
{
  int t = blockIdx.x * 256 + threadIdx.x;
  if (t >= D_ * D_) return;
  int n = t / D_, k = t - n * D_;
  out[t] = f2bf(in[(size_t)k * D_ + n]);
}

__global__ void k_zero32(uint32_t* __restrict__ p, int n)
{
  int t = blockIdx.x * 256 + threadIdx.x;
  if (t < n) p[t] = 0u;
}

// ---------- projection GEMM: head-major epilogue ----------
__global__ void k_proj(const uint16_t* __restrict__ xbf, const uint16_t* __restrict__ WT,
                       uint16_t* __restrict__ obf, float* __restrict__ of)
{
  const int m0 = blockIdx.y * 16, n0 = blockIdx.x * 16;
  v8f c = wmma_acc_bt(xbf, D_, WT, D_, m0, n0, D_);
  const int lane = threadIdx.x & 31, half = lane >> 4, l15 = lane & 15;
  const int n = n0 + l15;
  const int hh = n >> 6, dc = n & 63;
#pragma unroll
  for (int r = 0; r < 8; ++r) {
    int row = m0 + half * 8 + r;
    size_t e = ((size_t)hh * S_ + row) * DH_ + dc;
    obf[e] = f2bf(c[r]);
    if (of) of[e] = c[r];
  }
}

// ---------- per-(h,s) norms, kernel diagonal, preconditioner diagonal ----------
__global__ void k_norms(const uint16_t* __restrict__ qbf, const uint16_t* __restrict__ kbf,
                        const float* __restrict__ scal, const float* __restrict__ diag_scale,
                        const float* __restrict__ reg,
                        float* __restrict__ qn, float* __restrict__ kn,
                        float* __restrict__ diagp)
{
  int t = blockIdx.x * 256 + threadIdx.x;           // over H*S
  if (t >= H_ * S_) return;
  int h = t >> 11;
  const uint16_t* q = qbf + (size_t)t * DH_;
  const uint16_t* k = kbf + (size_t)t * DH_;
  float sq = 0.f, sk = 0.f, sd = 0.f;
#pragma unroll 4
  for (int i = 0; i < DH_; ++i) {
    float a = bf2f(q[i]), b = bf2f(k[i]);
    sq += a * a; sk += b * b;
    float d = a - b; sd += d * d;
  }
  qn[t] = sq; kn[t] = sk;
  float Kss = __expf(-fmaxf(sd, 0.f) * scal[0]);    // exact diag of K
  diagp[t] = log1pf(expf(Kss)) * diag_scale[h] + reg[0];
}

// ---------- U[h][s][p] = sum_r pe[s][r] * hp[h][r][p]  (tiny, f32 VALU) ----------
__global__ void k_U(const float* __restrict__ pe, const float* __restrict__ hp,
                    float* __restrict__ U)
{
  int t = blockIdx.x * 256 + threadIdx.x;           // over H*S*RK
  if (t >= H_ * S_ * RK_) return;
  int p = t & 15;
  int hs = t >> 4;
  int s = hs & (S_ - 1);
  int h = hs >> 11;
  const float* hpp = hp + (size_t)h * RK_ * RK_ + p;
  const float* pes = pe + (size_t)s * RK_;
  float sum = 0.f;
#pragma unroll
  for (int r = 0; r < RK_; ++r) sum += pes[r] * hpp[r * RK_];
  U[t] = sum;
}

// ---------- kernel matrix build: K[h][s][t] = exp(-max(dist,0)/(2 bw^2)), bf16 ----------
__global__ void k_buildK(const uint16_t* __restrict__ qbf, const uint16_t* __restrict__ kbf,
                         const float* __restrict__ qn, const float* __restrict__ kn,
                         const float* __restrict__ scal, uint16_t* __restrict__ Kb)
{
  const int h = blockIdx.z;
  const int m0 = blockIdx.y * 16, n0 = blockIdx.x * 16;
  const uint16_t* A  = qbf + (size_t)h * S_ * DH_;
  const uint16_t* BT = kbf + (size_t)h * S_ * DH_;
  v8f c = wmma_acc_bt(A, DH_, BT, DH_, m0, n0, DH_);
  const int lane = threadIdx.x & 31, half = lane >> 4, l15 = lane & 15;
  const int col = n0 + l15;
  const float inv2bw2 = scal[0];
  const float knv = kn[h * S_ + col];
  uint16_t* Kout = Kb + (size_t)h * S_ * S_;
#pragma unroll
  for (int r = 0; r < 8; ++r) {
    int row = m0 + half * 8 + r;
    float d = qn[h * S_ + row] + knv - 2.0f * c[r];
    d = fmaxf(d, 0.0f);
    Kout[(size_t)row * S_ + col] = f2bf(__expf(-d * inv2bw2));
  }
}

// ---------- Richardson GEMM: r = v - K*alpha - lam*alpha ----------
__global__ __launch_bounds__(256)
void k_kalpha_resid(const uint16_t* __restrict__ Kb, const uint16_t* __restrict__ aT,
                    const float* __restrict__ vf, const float* __restrict__ af,
                    const float* __restrict__ scal, float* __restrict__ rf)
{
  __shared__ __align__(16) uint16_t sB[64 * KCP_];
  const int h = blockIdx.z;
  v8f c[4];
  kalpha_acc64(Kb + (size_t)h * S_ * S_, aT + (size_t)h * DH_ * S_, sB, c);
  const int tid = threadIdx.x, lane = tid & 31, wv = tid >> 5;
  const int half = lane >> 4, l15 = lane & 15;
  const int m0 = blockIdx.y * 128 + wv * 16;
  const float lam = scal[1];
#pragma unroll
  for (int nt = 0; nt < 4; ++nt) {
    int col = nt * 16 + l15;
#pragma unroll
    for (int r = 0; r < 8; ++r) {
      int row = m0 + half * 8 + r;
      size_t e = ((size_t)h * S_ + row) * DH_ + col;
      rf[e] = vf[e] - c[nt][r] - lam * af[e];
    }
  }
}

// ---------- final K*alpha with concat-reshape epilogue (s, h*64+dh) bf16 ----------
__global__ __launch_bounds__(256)
void k_kalpha_out(const uint16_t* __restrict__ Kb, const uint16_t* __restrict__ aT,
                  uint16_t* __restrict__ tbf)
{
  __shared__ __align__(16) uint16_t sB[64 * KCP_];
  const int h = blockIdx.z;
  v8f c[4];
  kalpha_acc64(Kb + (size_t)h * S_ * S_, aT + (size_t)h * DH_ * S_, sB, c);
  const int tid = threadIdx.x, lane = tid & 31, wv = tid >> 5;
  const int half = lane >> 4, l15 = lane & 15;
  const int m0 = blockIdx.y * 128 + wv * 16;
#pragma unroll
  for (int nt = 0; nt < 4; ++nt) {
    int col = nt * 16 + l15;
#pragma unroll
    for (int r = 0; r < 8; ++r) {
      int row = m0 + half * 8 + r;
      tbf[(size_t)row * D_ + h * DH_ + col] = f2bf(c[nt][r]);
    }
  }
}

// ---------- ut_r[h][r][d] = sum_s U[h][s][r] * rbuf[h][s][d]  (f32) ----------
__global__ void k_utr(const float* __restrict__ U, const float* __restrict__ rf,
                      float* __restrict__ utr)
{
  int t = blockIdx.x * 256 + threadIdx.x;           // over H*RK*DH
  if (t >= H_ * RK_ * DH_) return;
  int d = t & 63;
  int hr = t >> 6;
  int r = hr & 15;
  int h = hr >> 4;
  const float* Uh = U  + (size_t)h * S_ * RK_ + r;
  const float* rh = rf + (size_t)h * S_ * DH_ + d;
  float s = 0.f;
  for (int ss = 0; ss < S_; ++ss) s += Uh[(size_t)ss * RK_] * rh[(size_t)ss * DH_];
  utr[t] = s;
}

// ---------- alpha += r*diag_p + U @ ut_r ; maintain alpha^T bf16 mirror ----------
__global__ void k_update(const float* __restrict__ U, const float* __restrict__ utr,
                         const float* __restrict__ rf, const float* __restrict__ diagp,
                         float* __restrict__ af, uint16_t* __restrict__ aT)
{
  int t = blockIdx.x * 256 + threadIdx.x;           // over H*S*DH
  if (t >= H_ * S_ * DH_) return;
  int d = t & 63;
  int hs = t >> 6;
  int s = hs & (S_ - 1);
  int h = hs >> 11;
  const float* Urow = U + (size_t)hs * RK_;
  const float* uc = utr + (size_t)h * RK_ * DH_ + d;
  float corr = 0.f;
#pragma unroll
  for (int p = 0; p < RK_; ++p) corr += Urow[p] * uc[p * DH_];
  float a = af[t] + rf[t] * diagp[hs] + corr;
  af[t] = a;
  aT[((size_t)h * DH_ + d) * S_ + s] = f2bf(a);
}

// ---------- final output GEMM: out(s, D) f32 = tbf @ Wo ----------
__global__ void k_gemm_out(const uint16_t* __restrict__ A, const uint16_t* __restrict__ BT,
                           float* __restrict__ C)
{
  const int m0 = blockIdx.y * 16, n0 = blockIdx.x * 16;
  v8f c = wmma_acc_bt(A, D_, BT, D_, m0, n0, D_);
  const int lane = threadIdx.x & 31, half = lane >> 4, l15 = lane & 15;
  const int col = n0 + l15;
#pragma unroll
  for (int r = 0; r < 8; ++r) {
    int row = m0 + half * 8 + r;
    C[(size_t)row * D_ + col] = c[r];
  }
}

// ---------------------------------------------------------------------------
extern "C" void kernel_launch(void* const* d_in, const int* in_sizes, int n_in,
                              void* d_out, int out_size, void* d_ws, size_t ws_size,
                              hipStream_t stream)
{
  const float* x    = (const float*)d_in[0];
  const float* Wq   = (const float*)d_in[1];
  const float* Wk   = (const float*)d_in[2];
  const float* Wv   = (const float*)d_in[3];
  const float* Wo   = (const float*)d_in[4];
  const float* bwp  = (const float*)d_in[5];
  const float* dsc  = (const float*)d_in[6];
  const float* pe   = (const float*)d_in[7];
  const float* hp   = (const float*)d_in[8];
  const float* reg  = (const float*)d_in[9];
  const float* lreg = (const float*)d_in[10];
  float* out = (float*)d_out;

  char* w = (char*)d_ws;
  size_t off = 0;
  auto take = [&](size_t bytes) -> void* {
    void* p = w + off;
    off = (off + bytes + 255) & ~(size_t)255;
    return p;
  };

  uint16_t* xbf = (uint16_t*)take((size_t)S_ * D_ * 2);
  uint16_t* WqT = (uint16_t*)take((size_t)D_ * D_ * 2);
  uint16_t* WkT = (uint16_t*)take((size_t)D_ * D_ * 2);
  uint16_t* WvT = (uint16_t*)take((size_t)D_ * D_ * 2);
  uint16_t* WoT = (uint16_t*)take((size_t)D_ * D_ * 2);
  uint16_t* qbf = (uint16_t*)take((size_t)H_ * S_ * DH_ * 2);
  uint16_t* kbf = (uint16_t*)take((size_t)H_ * S_ * DH_ * 2);
  uint16_t* vbf = (uint16_t*)take((size_t)H_ * S_ * DH_ * 2);
  float*    vf  = (float*)   take((size_t)H_ * S_ * DH_ * 4);
  uint16_t* Kb  = (uint16_t*)take((size_t)H_ * S_ * S_ * 2);   // 128 MB bf16, L2-resident
  float*    qn  = (float*)   take((size_t)H_ * S_ * 4);
  float*    kn  = (float*)   take((size_t)H_ * S_ * 4);
  float*    dgp = (float*)   take((size_t)H_ * S_ * 4);
  float*    U   = (float*)   take((size_t)H_ * S_ * RK_ * 4);
  float*    af  = (float*)   take((size_t)H_ * S_ * DH_ * 4);
  uint16_t* aT  = (uint16_t*)take((size_t)H_ * DH_ * S_ * 2);
  float*    rf  = (float*)   take((size_t)H_ * S_ * DH_ * 4);
  float*    utr = (float*)   take((size_t)H_ * RK_ * DH_ * 4);
  uint16_t* tbf = (uint16_t*)take((size_t)S_ * D_ * 2);
  float*    scal= (float*)   take(256);

  // scalars
  k_scalars<<<1, 32, 0, stream>>>(bwp, lreg, scal);

  // bf16 conversions (weights transposed for the B^T GEMM path)
  k_cvt <<<(S_ * D_ + 255) / 256, 256, 0, stream>>>(x, xbf, S_ * D_);
  k_cvtT<<<(D_ * D_ + 255) / 256, 256, 0, stream>>>(Wq, WqT);
  k_cvtT<<<(D_ * D_ + 255) / 256, 256, 0, stream>>>(Wk, WkT);
  k_cvtT<<<(D_ * D_ + 255) / 256, 256, 0, stream>>>(Wv, WvT);
  k_cvtT<<<(D_ * D_ + 255) / 256, 256, 0, stream>>>(Wo, WoT);

  // projections (WMMA), head-major outputs; v keeps an f32 mirror for the residual
  dim3 gp(D_ / 16, S_ / 16);
  k_proj<<<gp, 32, 0, stream>>>(xbf, WqT, qbf, (float*)nullptr);
  k_proj<<<gp, 32, 0, stream>>>(xbf, WkT, kbf, (float*)nullptr);
  k_proj<<<gp, 32, 0, stream>>>(xbf, WvT, vbf, vf);

  // norms / diag preconditioner / U
  k_norms<<<(H_ * S_ + 255) / 256, 256, 0, stream>>>(qbf, kbf, scal, dsc, reg, qn, kn, dgp);
  k_U<<<(H_ * S_ * RK_ + 255) / 256, 256, 0, stream>>>(pe, hp, U);

  // kernel matrix in bf16 (WMMA + fused exp epilogue)
  dim3 gK(S_ / 16, S_ / 16, H_);
  k_buildK<<<gK, 32, 0, stream>>>(qbf, kbf, qn, kn, scal, Kb);

  // alpha = 0 (f32 and bf16-transposed mirror)
  k_zero32<<<(H_ * S_ * DH_ + 255) / 256, 256, 0, stream>>>((uint32_t*)af, H_ * S_ * DH_);
  k_zero32<<<(H_ * DH_ * S_ / 2 + 255) / 256, 256, 0, stream>>>((uint32_t*)aT, H_ * DH_ * S_ / 2);

  // preconditioned Richardson iterations (async-LDS staged K*alpha GEMM)
  dim3 gA(1, S_ / 128, H_);
  for (int it = 0; it < ITERS_; ++it) {
    k_kalpha_resid<<<gA, 256, 0, stream>>>(Kb, aT, vf, af, scal, rf);
    k_utr<<<(H_ * RK_ * DH_ + 255) / 256, 256, 0, stream>>>(U, rf, utr);
    k_update<<<(H_ * S_ * DH_ + 255) / 256, 256, 0, stream>>>(U, utr, rf, dgp, af, aT);
  }

  // out = (K @ alpha) reshaped, then @ Wo
  k_kalpha_out<<<gA, 256, 0, stream>>>(Kb, aT, tbf);
  dim3 gO(D_ / 16, S_ / 16);
  k_gemm_out<<<gO, 32, 0, stream>>>(tbf, WoT, out);

  (void)in_sizes; (void)n_in; (void)out_size; (void)ws_size; (void)vbf;
}
